// PointNet2Generator_89300960018683
// MI455X (gfx1250) — compile-verified, matrix operand order checked by reading, not run on hardware
//
#include <hip/hip_runtime.h>
#include <hip/hip_bf16.h>
#include <stdint.h>

typedef __attribute__((ext_vector_type(16))) _Float16 v16h;
typedef __attribute__((ext_vector_type(8)))  float    v8f;

#define KNB 32
#define BDIM 4
#define NPTS 4096

// ---------------------------------------------------------------------------
// CDNA5 WMMA 16x16x32 f16 fragment geometry (ISA 7.12.2):
//   A: lane l holds row M = l&15; its 16 halves e=0..15 map to
//      K = 8*(l>>4) + { e (e<8) | e+8 (e>=8) }   -> two contiguous 8-runs
//   B: symmetric with N = l&15.
//   C/D: lane l, vgpr v -> M = v + 8*(l>>4), N = l&15.
// Block tile 128x64: 4 waves, each wave owns 2 m-subtiles x 4 n-subtiles
// (8 WMMAs / k-step). A comes straight from global (padded zero-filled
// 32-stride rows -> 4 unconditional global_load_b128 per fragment); B comes
// from pre-packed fragment-ordered f16 weights (1 contiguous 32B load per
// lane per n-tile, L2-resident). No LDS, no barriers.
// ---------------------------------------------------------------------------
__global__ __launch_bounds__(128) void gemm_wmma_kernel(
    const float* __restrict__ X, int ldx,
    const _Float16* __restrict__ Wp, int nta,
    const float* __restrict__ bias, float* __restrict__ Y, int ldy,
    int rows, int kTiles, int Cout, int doRelu) {
  const int t    = threadIdx.x;
  const int lane = t & 31;
  const int wave = t >> 5;
  const int bm = blockIdx.x * 128;
  const int bn = blockIdx.y * 64;

  int row0 = bm + wave * 16 + (lane & 15);
  int row1 = row0 + 64;
  if (row0 >= rows) row0 = rows - 1;            // clamp: masked at store
  if (row1 >= rows) row1 = rows - 1;
  const float* Xr0 = X + (size_t)row0 * ldx + ((lane >> 4) << 3);
  const float* Xr1 = X + (size_t)row1 * ldx + ((lane >> 4) << 3);
  const _Float16* WpL = Wp + ((size_t)(blockIdx.y * 4) * 32 + lane) * 16;
  const size_t ktStride = (size_t)nta * 32 * 16;   // halves per k-tile
  const size_t ntStride = 32 * 16;                 // halves per n-tile

  v8f acc0[4], acc1[4];
#pragma unroll
  for (int q = 0; q < 4; q++)
#pragma unroll
    for (int e = 0; e < 8; e++) { acc0[q][e] = 0.0f; acc1[q][e] = 0.0f; }

  for (int kt = 0; kt < kTiles; kt++) {
    const float* x0 = Xr0 + kt * 32;
    const float* x1 = Xr1 + kt * 32;
    float4 p0 = *(const float4*)(x0 + 0);
    float4 p1 = *(const float4*)(x0 + 4);
    float4 p2 = *(const float4*)(x0 + 16);
    float4 p3 = *(const float4*)(x0 + 20);
    float4 q0 = *(const float4*)(x1 + 0);
    float4 q1 = *(const float4*)(x1 + 4);
    float4 q2 = *(const float4*)(x1 + 16);
    float4 q3 = *(const float4*)(x1 + 20);
    v16h a0, a1;
    a0[0]  = (_Float16)p0.x; a0[1]  = (_Float16)p0.y;
    a0[2]  = (_Float16)p0.z; a0[3]  = (_Float16)p0.w;
    a0[4]  = (_Float16)p1.x; a0[5]  = (_Float16)p1.y;
    a0[6]  = (_Float16)p1.z; a0[7]  = (_Float16)p1.w;
    a0[8]  = (_Float16)p2.x; a0[9]  = (_Float16)p2.y;
    a0[10] = (_Float16)p2.z; a0[11] = (_Float16)p2.w;
    a0[12] = (_Float16)p3.x; a0[13] = (_Float16)p3.y;
    a0[14] = (_Float16)p3.z; a0[15] = (_Float16)p3.w;
    a1[0]  = (_Float16)q0.x; a1[1]  = (_Float16)q0.y;
    a1[2]  = (_Float16)q0.z; a1[3]  = (_Float16)q0.w;
    a1[4]  = (_Float16)q1.x; a1[5]  = (_Float16)q1.y;
    a1[6]  = (_Float16)q1.z; a1[7]  = (_Float16)q1.w;
    a1[8]  = (_Float16)q2.x; a1[9]  = (_Float16)q2.y;
    a1[10] = (_Float16)q2.z; a1[11] = (_Float16)q2.w;
    a1[12] = (_Float16)q3.x; a1[13] = (_Float16)q3.y;
    a1[14] = (_Float16)q3.z; a1[15] = (_Float16)q3.w;
    const _Float16* wp = WpL + (size_t)kt * ktStride;
#pragma unroll
    for (int nt = 0; nt < 4; nt++) {
      v16h bfrag = *(const v16h*)(wp + (size_t)nt * ntStride);
      acc0[nt] = __builtin_amdgcn_wmma_f32_16x16x32_f16(
          false, a0, false, bfrag, (short)0, acc0[nt], false, false);
      acc1[nt] = __builtin_amdgcn_wmma_f32_16x16x32_f16(
          false, a1, false, bfrag, (short)0, acc1[nt], false, false);
    }
  }

#pragma unroll
  for (int nt = 0; nt < 4; nt++) {
#pragma unroll
    for (int vi = 0; vi < 8; vi++) {
      int c = bn + nt * 16 + (lane & 15);
      if (c < ldy) {
        float bv = (bias && c < Cout) ? bias[c] : 0.0f;
        int r0 = bm + wave * 16 + vi + ((lane >> 4) << 3);
        if (r0 < rows) {
          float val = 0.0f;
          if (c < Cout) {
            val = acc0[nt][vi] + bv;
            if (doRelu) val = fmaxf(val, 0.0f);
          }
          Y[(size_t)r0 * ldy + c] = val;       // zero-fills the pad too
        }
        int r1 = r0 + 64;
        if (r1 < rows) {
          float val = 0.0f;
          if (c < Cout) {
            val = acc1[nt][vi] + bv;
            if (doRelu) val = fmaxf(val, 0.0f);
          }
          Y[(size_t)r1 * ldy + c] = val;
        }
      }
    }
  }
}

// Pack f32 weights W[Cin,Cout] -> f16 fragment-ordered, zero-padded:
// Wp[((kt*nta + nt)*32 + lane)*16 + e]
__global__ void pack_w_kernel(const float* __restrict__ W, _Float16* __restrict__ Wp,
                              int Cin, int Cout, int kTiles, int nta) {
  size_t total = (size_t)kTiles * nta * 512;
  size_t gid = (size_t)blockIdx.x * 256 + threadIdx.x;
  if (gid >= total) return;
  int e    = (int)(gid & 15);
  int lane = (int)((gid >> 4) & 31);
  size_t r = gid >> 9;
  int nt = (int)(r % nta);
  int kt = (int)(r / nta);
  int k  = kt * 32 + ((lane >> 4) << 3) + (e < 8 ? e : e + 8);
  int nn = nt * 16 + (lane & 15);
  float v = (k < Cin && nn < Cout) ? W[(size_t)k * Cout + nn] : 0.0f;
  Wp[gid] = (_Float16)v;
}

// ---------------------------------------------------------------------------
// [B,3,N] -> [B,N,3]
// ---------------------------------------------------------------------------
__global__ void transpose_in_kernel(const float* __restrict__ data,
                                    float* __restrict__ pos0, int N) {
  int gid = blockIdx.x * 256 + threadIdx.x;
  if (gid >= BDIM * N * 3) return;
  int b = gid / (N * 3);
  int r = gid - b * N * 3;
  int i = r / 3;
  int c = r - i * 3;
  pos0[gid] = data[(size_t)b * 3 * N + (size_t)c * N + i];
}

// ---------------------------------------------------------------------------
// FPS: one block/batch; wave32 shfl_xor reductions (2 barriers per iter).
// Tie-break: first occurrence of max (matches jnp.argmax).
// ---------------------------------------------------------------------------
#define FPS_T 1024
__global__ __launch_bounds__(FPS_T) void fps_kernel(
    const float* __restrict__ pos, int Np, int n, int* __restrict__ idxOut) {
  __shared__ float sd[NPTS];
  __shared__ float wv[FPS_T / 32];
  __shared__ int   wi[FPS_T / 32];
  __shared__ int   s_last;
  const int b = blockIdx.x;
  const float* P = pos + (size_t)b * Np * 3;
  int* idx = idxOut + (size_t)b * n;
  const int t = threadIdx.x;
  for (int i = t; i < Np; i += FPS_T) sd[i] = 1e10f;
  if (t == 0) { idx[0] = 0; s_last = 0; }
  __syncthreads();
  for (int s = 1; s < n; s++) {
    int last = s_last;
    float lx = P[last * 3], ly = P[last * 3 + 1], lz = P[last * 3 + 2];
    float bestv = -1.0f; int besti = 0;
    for (int i = t; i < Np; i += FPS_T) {
      float dx = P[i * 3] - lx, dy = P[i * 3 + 1] - ly, dz = P[i * 3 + 2] - lz;
      float cur = fminf(sd[i], dx * dx + dy * dy + dz * dz);
      sd[i] = cur;
      if (cur > bestv) { bestv = cur; besti = i; }
    }
#pragma unroll
    for (int m = 16; m; m >>= 1) {
      float v2 = __shfl_xor(bestv, m, 32);
      int   i2 = __shfl_xor(besti, m, 32);
      if (v2 > bestv || (v2 == bestv && i2 < besti)) { bestv = v2; besti = i2; }
    }
    if ((t & 31) == 0) { wv[t >> 5] = bestv; wi[t >> 5] = besti; }
    __syncthreads();
    if (t < 32) {
      bestv = wv[t]; besti = wi[t];
#pragma unroll
      for (int m = 16; m; m >>= 1) {
        float v2 = __shfl_xor(bestv, m, 32);
        int   i2 = __shfl_xor(besti, m, 32);
        if (v2 > bestv || (v2 == bestv && i2 < besti)) { bestv = v2; besti = i2; }
      }
      if (t == 0) { s_last = besti; idx[s] = besti; }
    }
    __syncthreads();
  }
}

__global__ void gather_pos_kernel(const float* __restrict__ posIn,
                                  const int* __restrict__ idx,
                                  float* __restrict__ posOut, int NpIn, int n) {
  int b = blockIdx.y;
  int i = blockIdx.x * 256 + threadIdx.x;
  if (i >= n) return;
  int j = idx[(size_t)b * n + i];
  const float* s = posIn + ((size_t)b * NpIn + j) * 3;
  float* d = posOut + ((size_t)b * n + i) * 3;
  d[0] = s[0]; d[1] = s[1]; d[2] = s[2];
}

__global__ __launch_bounds__(128) void ballquery_kernel(
    const float* __restrict__ ctr, const float* __restrict__ pos,
    int n, int Np, float r2, int* __restrict__ nb, int* __restrict__ valid) {
  int b = blockIdx.y;
  int i = blockIdx.x * 128 + threadIdx.x;
  if (i >= n) return;
  const float* C = ctr + ((size_t)b * n + i) * 3;
  const float* P = pos + (size_t)b * Np * 3;
  float cx = C[0], cy = C[1], cz = C[2];
  float dk[KNB]; int ik[KNB];
  for (int k = 0; k < KNB; k++) { dk[k] = 1e30f; ik[k] = 0; }
  for (int j = 0; j < Np; j++) {
    float dx = P[j * 3] - cx, dy = P[j * 3 + 1] - cy, dz = P[j * 3 + 2] - cz;
    float d = dx * dx + dy * dy + dz * dz;
    if (d < dk[KNB - 1]) {      // strict <  => jax top_k tie-break
      int p = KNB - 1;
      while (p > 0 && dk[p - 1] > d) { dk[p] = dk[p - 1]; ik[p] = ik[p - 1]; p--; }
      dk[p] = d; ik[p] = j;
    }
  }
  size_t base = ((size_t)b * n + i) * KNB;
  for (int k = 0; k < KNB; k++) {
    nb[base + k] = ik[k];
    valid[base + k] = (dk[k] <= r2) ? 1 : 0;
  }
}

__global__ void knn_kernel(const float* __restrict__ posF,
                           const float* __restrict__ posC,
                           int Nf, int Nc, int* __restrict__ nb3,
                           float* __restrict__ w3) {
  int b = blockIdx.y;
  int i = blockIdx.x * 256 + threadIdx.x;
  if (i >= Nf) return;
  const float* F = posF + ((size_t)b * Nf + i) * 3;
  const float* P = posC + (size_t)b * Nc * 3;
  float fx = F[0], fy = F[1], fz = F[2];
  float d0 = 1e30f, d1 = 1e30f, d2 = 1e30f;
  int i0 = 0, i1 = 0, i2 = 0;
  for (int j = 0; j < Nc; j++) {
    float dx = P[j * 3] - fx, dy = P[j * 3 + 1] - fy, dz = P[j * 3 + 2] - fz;
    float d = dx * dx + dy * dy + dz * dz;
    if (d < d0)      { d2 = d1; i2 = i1; d1 = d0; i1 = i0; d0 = d; i0 = j; }
    else if (d < d1) { d2 = d1; i2 = i1; d1 = d;  i1 = j; }
    else if (d < d2) { d2 = d;  i2 = j; }
  }
  size_t base = ((size_t)b * Nf + i) * 3;
  nb3[base] = i0; nb3[base + 1] = i1; nb3[base + 2] = i2;
  w3[base]     = 1.0f / fmaxf(d0, 1e-16f);
  w3[base + 1] = 1.0f / fmaxf(d1, 1e-16f);
  w3[base + 2] = 1.0f / fmaxf(d2, 1e-16f);
}

// ---------------------------------------------------------------------------
// Per-batch kernels (pointers pre-offset for the batch)
// ---------------------------------------------------------------------------
__global__ void edge_kernel(const float* __restrict__ x, int ldX,
                            const float* __restrict__ pos,
                            const float* __restrict__ ctr,
                            const int* __restrict__ nb,
                            float* __restrict__ out, int n, int Cx, int ldOut) {
  size_t total = (size_t)n * KNB * ldOut;
  size_t gid = (size_t)blockIdx.x * 256 + threadIdx.x;
  if (gid >= total) return;
  int c = (int)(gid % ldOut);
  size_t r = gid / ldOut;               // r = i*K + k
  int i = (int)(r >> 5);
  float v = 0.0f;
  if (c < Cx) {
    int j = nb[r];
    v = x[(size_t)j * ldX + c];
  } else if (c < Cx + 3) {
    int j = nb[r];
    v = pos[(size_t)j * 3 + (c - Cx)] - ctr[(size_t)i * 3 + (c - Cx)];
  }
  out[gid] = v;
}

__global__ void maxpool_kernel(const float* __restrict__ h, int ldh,
                               const int* __restrict__ valid,
                               float* __restrict__ out, int n, int C) {
  size_t total = (size_t)n * C;
  size_t gid = (size_t)blockIdx.x * 256 + threadIdx.x;
  if (gid >= total) return;
  int c = (int)(gid % C);
  int i = (int)(gid / C);
  float m = 0.0f;  // post-ReLU >= 0 and self is always valid => exact
  for (int k = 0; k < KNB; k++)
    if (valid[(size_t)i * KNB + k]) m = fmaxf(m, h[((size_t)i * KNB + k) * ldh + c]);
  out[gid] = m;
}

__global__ void fpconcat_kernel(const float* __restrict__ xc,
                                const int* __restrict__ nb3,
                                const float* __restrict__ w3,
                                const float* __restrict__ xs, int ldxs,
                                const float* __restrict__ posF,
                                float* __restrict__ out,
                                int Nf, int Cc, int Cs, int ldOut) {
  size_t total = (size_t)Nf * ldOut;
  size_t gid = (size_t)blockIdx.x * 256 + threadIdx.x;
  if (gid >= total) return;
  int c = (int)(gid % ldOut);
  int i = (int)(gid / ldOut);
  float v = 0.0f;
  if (c < Cc) {
    const int* np = nb3 + (size_t)i * 3;
    const float* wp = w3 + (size_t)i * 3;
    float wsum = wp[0] + wp[1] + wp[2];
    v = (wp[0] * xc[(size_t)np[0] * Cc + c] +
         wp[1] * xc[(size_t)np[1] * Cc + c] +
         wp[2] * xc[(size_t)np[2] * Cc + c]) / wsum;
  } else if (c < Cc + Cs) {
    v = xs[(size_t)i * ldxs + (c - Cc)];
  } else if (c < Cc + Cs + 3) {
    v = posF[(size_t)i * 3 + (c - Cc - Cs)];
  }
  out[gid] = v;
}

__global__ void residual_out_kernel(const float* __restrict__ h, int ldh,
                                    const float* __restrict__ data,
                                    float* __restrict__ out, int N) {
  int gid = blockIdx.x * 256 + threadIdx.x;
  if (gid >= 3 * N) return;
  int c = gid / N;
  int i = gid - c * N;
  out[gid] = h[(size_t)i * ldh + c] + data[gid];
}

// ---------------------------------------------------------------------------
// Host side
// ---------------------------------------------------------------------------
struct PackedLayer {
  _Float16* wp;
  const float* bias;
  int wIdx, cin, cout, kt, nta, ldOut;
};

static void launch_gemm(const float* X, int ldx, const PackedLayer& L,
                        float* Y, int rows, int relu, hipStream_t stream) {
  dim3 g((rows + 127) / 128, (L.cout + 63) / 64);
  gemm_wmma_kernel<<<g, 128, 0, stream>>>(X, ldx, L.wp, L.nta, L.bias, Y,
                                          L.ldOut, rows, L.kt, L.cout, relu);
}

static const float* run_chain(const float* src, int ldSrc, int rows,
                              const PackedLayer* Ls, int nl,
                              float* bufA, float* bufB, float* finalDst,
                              hipStream_t stream, int* ldOutRet) {
  const float* cur = src;
  int ldc = ldSrc;
  for (int j = 0; j < nl; j++) {
    float* dst = (j == nl - 1 && finalDst) ? finalDst
                                           : ((cur == bufA) ? bufB : bufA);
    launch_gemm(cur, ldc, Ls[j], dst, rows, 1, stream);
    cur = dst;
    ldc = Ls[j].ldOut;
  }
  *ldOutRet = ldc;
  return cur;
}

extern "C" void kernel_launch(void* const* d_in, const int* in_sizes, int n_in,
                              void* d_out, int out_size, void* d_ws, size_t ws_size,
                              hipStream_t stream) {
  (void)in_sizes; (void)out_size;
  if (n_in < 63) return;
  const int N = NPTS;
  const int S[5] = {4096, 2868, 2008, 1406, 985};
  const float R2[4] = {0.025f * 0.025f, 0.05f * 0.05f, 0.1f * 0.1f, 0.2f * 0.2f};
  const int xC[5] = {3, 64, 128, 256, 512};

  const int saNL[4] = {3, 4, 4, 4};
  const int saDims[4][5] = {{6, 32, 32, 64, 0},
                            {67, 64, 64, 64, 128},
                            {131, 128, 128, 128, 256},
                            {259, 256, 256, 256, 512}};
  const int saWi[4][4] = {{1, 3, 5, 0}, {7, 9, 11, 13}, {15, 17, 19, 21}, {23, 25, 27, 29}};
  const int saBi[4][4] = {{2, 4, 6, 0}, {8, 10, 12, 14}, {16, 18, 20, 22}, {24, 26, 28, 30}};
  const int fpNL[4] = {4, 4, 2, 4};  // fp4, fp3, fp2, fp1
  const int fpDims[4][5] = {{771, 512, 256, 256, 256},
                            {387, 256, 256, 256, 256},
                            {323, 256, 256, 0, 0},
                            {262, 256, 128, 128, 128}};
  const int fpWi[4][4] = {{31, 33, 35, 37}, {39, 41, 43, 45}, {47, 49, 0, 0}, {51, 53, 55, 57}};
  const int fpBi[4][4] = {{32, 34, 36, 38}, {40, 42, 44, 46}, {48, 50, 0, 0}, {52, 54, 56, 58}};
  const int fcWi[4] = {59, 60, 61, 62};
  const int fcDims[5] = {128, 64, 32, 16, 3};

  auto in = [&](int i) { return (const float*)d_in[i]; };
  auto pad32 = [](int x) { return (x + 31) & ~31; };
  const float* data = in(0);
  float* out = (float*)d_out;

  // ---- deterministic bump allocator over d_ws ----
  char* ws = (char*)d_ws;
  size_t off = 0;
  auto alloc = [&](size_t bytes) -> size_t {
    size_t o = off;
    off += (bytes + 255) & ~(size_t)255;
    return o;
  };
  size_t posOff[5], fidxOff[4], nbOff[4], validOff[4], nb3Off[4], w3Off[4];
  for (int l = 0; l < 5; l++) posOff[l] = alloc((size_t)BDIM * S[l] * 3 * sizeof(float));
  for (int l = 0; l < 4; l++) fidxOff[l] = alloc((size_t)BDIM * S[l + 1] * sizeof(int));
  for (int l = 0; l < 4; l++) nbOff[l] = alloc((size_t)BDIM * S[l + 1] * KNB * sizeof(int));
  for (int l = 0; l < 4; l++) validOff[l] = alloc((size_t)BDIM * S[l + 1] * KNB * sizeof(int));
  for (int f = 0; f < 4; f++) nb3Off[f] = alloc((size_t)BDIM * S[3 - f] * 3 * sizeof(int));
  for (int f = 0; f < 4; f++) w3Off[f] = alloc((size_t)BDIM * S[3 - f] * 3 * sizeof(float));
  size_t xlevOff[5];
  xlevOff[0] = 0;  // x0 aliases pos0
  for (int l = 1; l < 5; l++) xlevOff[l] = alloc((size_t)S[l] * xC[l] * sizeof(float));
  size_t yOff[4];
  const int yC[4] = {256, 256, 256, 128};
  for (int f = 0; f < 4; f++) yOff[f] = alloc((size_t)S[3 - f] * yC[f] * sizeof(float));
  const size_t bufElems = (size_t)S[4] * KNB * 512;  // max rows*ld over all layers
  size_t bufAOff = alloc(bufElems * sizeof(float));
  size_t bufBOff = alloc(bufElems * sizeof(float));

  // packed weight allocations
  PackedLayer saL[4][4], fpL[4][4], fcL[4];
  auto mkLayer = [&](int wIdx, int bIdx, int cin, int cout) -> PackedLayer {
    PackedLayer L;
    L.wIdx = wIdx; L.cin = cin; L.cout = cout;
    L.kt = (cin + 31) / 32;
    L.nta = ((cout + 63) / 64) * 4;
    L.ldOut = pad32(cout);
    L.wp = (_Float16*)(ws + alloc((size_t)L.kt * L.nta * 512 * sizeof(_Float16)));
    L.bias = (bIdx >= 0) ? in(bIdx) : nullptr;
    return L;
  };
  for (int l = 0; l < 4; l++)
    for (int j = 0; j < saNL[l]; j++)
      saL[l][j] = mkLayer(saWi[l][j], saBi[l][j], saDims[l][j], saDims[l][j + 1]);
  for (int f = 0; f < 4; f++)
    for (int j = 0; j < fpNL[f]; j++)
      fpL[f][j] = mkLayer(fpWi[f][j], fpBi[f][j], fpDims[f][j], fpDims[f][j + 1]);
  for (int j = 0; j < 4; j++) fcL[j] = mkLayer(fcWi[j], -1, fcDims[j], fcDims[j + 1]);

  if (off > ws_size) return;  // workspace too small: deterministic no-op

  float* posW[5]; for (int l = 0; l < 5; l++) posW[l] = (float*)(ws + posOff[l]);
  int* fidxW[4];  for (int l = 0; l < 4; l++) fidxW[l] = (int*)(ws + fidxOff[l]);
  int* nbW[4];    for (int l = 0; l < 4; l++) nbW[l] = (int*)(ws + nbOff[l]);
  int* validW[4]; for (int l = 0; l < 4; l++) validW[l] = (int*)(ws + validOff[l]);
  int* nb3W[4];   for (int f = 0; f < 4; f++) nb3W[f] = (int*)(ws + nb3Off[f]);
  float* w3W[4];  for (int f = 0; f < 4; f++) w3W[f] = (float*)(ws + w3Off[f]);
  float* xlevW[5]; for (int l = 1; l < 5; l++) xlevW[l] = (float*)(ws + xlevOff[l]);
  float* yW[4];   for (int f = 0; f < 4; f++) yW[f] = (float*)(ws + yOff[f]);
  float* bufA = (float*)(ws + bufAOff);
  float* bufB = (float*)(ws + bufBOff);

  // ---- Phase 0: weight packing (layer constants, cached in L2) ----
  auto packOne = [&](const PackedLayer& L) {
    size_t total = (size_t)L.kt * L.nta * 512;
    pack_w_kernel<<<(unsigned)((total + 255) / 256), 256, 0, stream>>>(
        in(L.wIdx), L.wp, L.cin, L.cout, L.kt, L.nta);
  };
  for (int l = 0; l < 4; l++) for (int j = 0; j < saNL[l]; j++) packOne(saL[l][j]);
  for (int f = 0; f < 4; f++) for (int j = 0; j < fpNL[f]; j++) packOne(fpL[f][j]);
  for (int j = 0; j < 4; j++) packOne(fcL[j]);

  // ---- Phase 1: geometry pyramid for all batches (feature-independent) ----
  transpose_in_kernel<<<(BDIM * N * 3 + 255) / 256, 256, 0, stream>>>(data, posW[0], N);
  for (int l = 0; l < 4; l++) {
    fps_kernel<<<BDIM, FPS_T, 0, stream>>>(posW[l], S[l], S[l + 1], fidxW[l]);
    gather_pos_kernel<<<dim3((S[l + 1] + 255) / 256, BDIM), 256, 0, stream>>>(
        posW[l], fidxW[l], posW[l + 1], S[l], S[l + 1]);
    ballquery_kernel<<<dim3((S[l + 1] + 127) / 128, BDIM), 128, 0, stream>>>(
        posW[l + 1], posW[l], S[l + 1], S[l], R2[l], nbW[l], validW[l]);
  }
  for (int f = 0; f < 4; f++) {
    int lf = 3 - f, lc = 4 - f;
    knn_kernel<<<dim3((S[lf] + 255) / 256, BDIM), 256, 0, stream>>>(
        posW[lf], posW[lc], S[lf], S[lc], nb3W[f], w3W[f]);
  }

  // ---- Phase 2: MLP pipeline per batch (reuses big buffers) ----
  for (int b = 0; b < BDIM; b++) {
    const float* pos_b[5];
    for (int l = 0; l < 5; l++) pos_b[l] = posW[l] + (size_t)b * S[l] * 3;
    const float* x_b[5];
    x_b[0] = pos_b[0];

    // SA levels
    for (int l = 0; l < 4; l++) {
      int n = S[l + 1];
      int rows = n * KNB;
      int Cx = xC[l];
      int ldX = (l == 0) ? 3 : xC[l];
      int Cout = saDims[l][saNL[l]];
      int ldE = pad32(Cx + 3);
      const int* nb_b = nbW[l] + (size_t)b * n * KNB;
      const int* valid_b = validW[l] + (size_t)b * n * KNB;
      size_t etotal = (size_t)rows * ldE;
      edge_kernel<<<(unsigned)((etotal + 255) / 256), 256, 0, stream>>>(
          x_b[l], ldX, pos_b[l], pos_b[l + 1], nb_b, bufA, n, Cx, ldE);
      int ldh = 0;
      const float* res = run_chain(bufA, ldE, rows, saL[l], saNL[l],
                                   bufA, bufB, nullptr, stream, &ldh);
      size_t ptotal = (size_t)n * Cout;
      maxpool_kernel<<<(unsigned)((ptotal + 255) / 256), 256, 0, stream>>>(
          res, ldh, valid_b, xlevW[l + 1], n, Cout);
      x_b[l + 1] = xlevW[l + 1];
    }

    // FP levels
    const float* xc = x_b[4];
    int Cc = 512;
    for (int f = 0; f < 4; f++) {
      int lf = 3 - f;
      int Nf = S[lf];
      int Cs = xC[lf];
      int ldxs = (lf == 0) ? 3 : xC[lf];
      const float* xs = x_b[lf];
      const int* nb3_b = nb3W[f] + (size_t)b * Nf * 3;
      const float* w3_b = w3W[f] + (size_t)b * Nf * 3;
      int ldC = pad32(Cc + Cs + 3);
      size_t ctotal = (size_t)Nf * ldC;
      fpconcat_kernel<<<(unsigned)((ctotal + 255) / 256), 256, 0, stream>>>(
          xc, nb3_b, w3_b, xs, ldxs, pos_b[lf], bufA, Nf, Cc, Cs, ldC);
      int ldy = 0;
      run_chain(bufA, ldC, Nf, fpL[f], fpNL[f], bufA, bufB, yW[f], stream, &ldy);
      xc = yW[f];
      Cc = fpDims[f][fpNL[f]];
    }

    // FC head: relu only after first layer, no biases
    launch_gemm(yW[3], 128, fcL[0], bufA, S[0], 1, stream);
    launch_gemm(bufA, fcL[0].ldOut, fcL[1], bufB, S[0], 0, stream);
    launch_gemm(bufB, fcL[1].ldOut, fcL[2], bufA, S[0], 0, stream);
    launch_gemm(bufA, fcL[2].ldOut, fcL[3], bufB, S[0], 0, stream);
    residual_out_kernel<<<(3 * N + 255) / 256, 256, 0, stream>>>(
        bufB, fcL[3].ldOut, data + (size_t)b * 3 * N, out + (size_t)b * 3 * N, N);
  }
}
